// MPN_Featurizer_11802570129437
// MI455X (gfx1250) — matrix operand order repaired and checked by reading, not compiled, not run on hardware
//
#include <hip/hip_runtime.h>
#include <cstdint>
#include <cstddef>

// ---------------------------------------------------------------------------
// MI455X (gfx1250) fused DMPNN featurizer.
//   * wave32: each wave owns a 16-edge (or 16-node) tile
//   * GEMMs via v_wmma_f32_16x16x32_f16 (f32 accumulate)
//   * segment_sum via global_atomic_add_f32 into an L2-resident [N,32] buffer
//   * edge update is in-place: rev(e) = e^1 is always inside the wave's tile
//   * full tiles take a branchless path; epilogue staged through LDS so all
//     global traffic in the hot path is b128, ordered by s_wait_dscnt
// ---------------------------------------------------------------------------

typedef __attribute__((ext_vector_type(16))) _Float16 v16h;
typedef __attribute__((ext_vector_type(8)))  float    v8f;

#define WMMA_F16(a, b, c) \
  __builtin_amdgcn_wmma_f32_16x16x32_f16(false, (a), false, (b), (short)0, (c), false, false)

// LDS store->load fence within a wave: DS ops are in-order per wave in HW; the
// explicit wait + memory clobber also stops compiler reordering.
#define LDS_FENCE() asm volatile("s_wait_dscnt 0x0" ::: "memory")

// ---------------- concrete dropout (deterministic hash RNG) ----------------

__device__ __forceinline__ uint32_t mix32(uint32_t x) {
  x ^= x >> 16; x *= 0x7FEB352Du;
  x ^= x >> 15; x *= 0x846CA68Bu;
  x ^= x >> 16; return x;
}

struct CDParams { float logit_p; float inv_1mp; };

__device__ __forceinline__ CDParams cd_params(const float* __restrict__ p_logit) {
  const float eps = 1e-7f;
  float pl = p_logit[0];
  float p  = 1.0f / (1.0f + __expf(-pl));
  CDParams q;
  q.logit_p = __logf(p + eps) - __logf(1.0f - p + eps);
  q.inv_1mp = 1.0f / (1.0f - p);
  return q;
}

__device__ __forceinline__ float cdrop(float x, CDParams q, uint32_t seed, uint32_t idx) {
  const float eps = 1e-7f;
  uint32_t h = mix32(idx * 0x9E3779B1u + seed * 0x85EBCA77u + 0x165667B1u);
  float u = (float)(h >> 8) * (1.0f / 16777216.0f);          // [0,1)
  float g = (q.logit_p + __logf(u + eps) - __logf(1.0f - u + eps)) * 10.0f; // /temp(0.1)
  float drop = 1.0f / (1.0f + __expf(-g));
  return x * (1.0f - drop) * q.inv_1mp;
}

// ---------------- small helper kernels ----------------

__global__ void convert_weights(const float* __restrict__ Wi,   // [32,20]
                                const float* __restrict__ We,   // [32,32]
                                const float* __restrict__ Wl,   // [64,47]
                                _Float16* __restrict__ Wih,     // [32,32] padded
                                _Float16* __restrict__ Weh,     // [32,32]
                                _Float16* __restrict__ Wlh) {   // [64,64] padded
  for (int i = threadIdx.x; i < 32 * 32; i += blockDim.x) {
    int o = i >> 5, k = i & 31;
    Wih[i] = (_Float16)(k < 20 ? Wi[o * 20 + k] : 0.0f);
  }
  for (int i = threadIdx.x; i < 32 * 32; i += blockDim.x)
    Weh[i] = (_Float16)We[i];
  for (int i = threadIdx.x; i < 64 * 64; i += blockDim.x) {
    int o = i >> 6, k = i & 63;
    Wlh[i] = (_Float16)(k < 47 ? Wl[o * 47 + k] : 0.0f);
  }
}

__global__ void zero_f32(float* __restrict__ p, int n) {
  int i = blockIdx.x * blockDim.x + threadIdx.x;
  if (i < n) p[i] = 0.0f;
}

// segment_sum(et, dst) -> neigh[N,32]; neigh (12.8 MB) is L2-resident.
__global__ __launch_bounds__(256)
void scatter_add32(const float* __restrict__ et, const int* __restrict__ dst,
                   float* __restrict__ neigh, int E) {
  int idx = blockIdx.x * blockDim.x + threadIdx.x;   // E*8 threads, 4 floats each
  if (idx >= E * 8) return;
  int e = idx >> 3, g = (idx & 7) << 2;
  float4 v = *(const float4*)(et + (size_t)e * 32 + g);
  float* nr = neigh + (size_t)dst[e] * 32 + g;
  __hip_atomic_fetch_add(nr + 0, v.x, __ATOMIC_RELAXED, __HIP_MEMORY_SCOPE_AGENT);
  __hip_atomic_fetch_add(nr + 1, v.y, __ATOMIC_RELAXED, __HIP_MEMORY_SCOPE_AGENT);
  __hip_atomic_fetch_add(nr + 2, v.z, __ATOMIC_RELAXED, __HIP_MEMORY_SCOPE_AGENT);
  __hip_atomic_fetch_add(nr + 3, v.w, __ATOMIC_RELAXED, __HIP_MEMORY_SCOPE_AGENT);
}

// ---------------- init message: im = relu(W_init @ CD(concat(h_src, e_w))) ----------------

__global__ __launch_bounds__(256)
void init_msg(const float* __restrict__ nf, const float* __restrict__ ef,
              const int* __restrict__ src, const _Float16* __restrict__ Wih,
              const float* __restrict__ p_logit,
              float* __restrict__ e0, float* __restrict__ et,
              int E, uint32_t seed) {
  __shared__ float smem[8 * 16 * 32];                 // 2 KB per wave
  int wave = (blockIdx.x * blockDim.x + threadIdx.x) >> 5;
  int ntiles = (E + 15) >> 4;
  if (wave >= ntiles) return;
  int lane = threadIdx.x & 31;
  float* tile = smem + ((threadIdx.x >> 5) & 7) * 512;
  int row = lane & 15, hi = lane >> 4;
  int base = wave << 4;
  int e = base + row;
  bool full = (base + 16 <= E);                       // wave-uniform
  bool eok = full || (e < E);
  int s = eok ? src[e] : 0;
  CDParams q = cd_params(p_logit);
  int ko = hi << 4;                      // A layout: K = ko..ko+15

  v16h a;
#pragma unroll
  for (int j = 0; j < 16; ++j) {
    int k = ko + j;
    float x = 0.0f;
    if (eok && k < 20) {
      x = (k < 15) ? nf[(size_t)s * 15 + k] : ef[(size_t)e * 5 + (k - 15)];
      x = cdrop(x, q, seed, (uint32_t)e * 32u + (uint32_t)k);
    }
    a[j] = (_Float16)x;
  }

  int n = row, kb = hi << 4;             // B layout: lane's 16 contiguous K values
  v16h b0 = *(const v16h*)(Wih + (n)      * 32 + kb);
  v16h b1 = *(const v16h*)(Wih + (n + 16) * 32 + kb);
  v8f c0 = {}, c1 = {};
  c0 = WMMA_F16(a, b0, c0);
  c1 = WMMA_F16(a, b1, c1);

  if (full) {
    // stage relu'd tile in LDS, then write both buffers as coalesced b128
#pragma unroll
    for (int r = 0; r < 8; ++r) {
      int M = r + (hi << 3);
      tile[M * 32 + n]      = fmaxf(c0[r], 0.0f);
      tile[M * 32 + 16 + n] = fmaxf(c1[r], 0.0f);
    }
    LDS_FENCE();
    int orow = lane >> 1, half = lane & 1;
    const float* tr = tile + orow * 32 + half * 16;
    size_t off = (size_t)(base + orow) * 32 + (half << 4);
#pragma unroll
    for (int u = 0; u < 4; ++u) {
      float4 v = *(const float4*)(tr + u * 4);
      *(float4*)(e0 + off + u * 4) = v;
      *(float4*)(et + off + u * 4) = v;
    }
  } else {
#pragma unroll
    for (int r = 0; r < 8; ++r) {
      int M = r + (hi << 3);
      int ee = base + M;
      if (ee < E) {
        size_t off = (size_t)ee * 32 + n;
        float v0 = fmaxf(c0[r], 0.0f);
        float v1 = fmaxf(c1[r], 0.0f);
        e0[off]      = v0;  et[off]      = v0;
        e0[off + 16] = v1;  et[off + 16] = v1;
      }
    }
  }
}

// ---------------- edge update: et = relu(e0 + CD(neigh[src] - et[rev]) @ W.T), in place ----------------

__global__ __launch_bounds__(256)
void edge_update(const int* __restrict__ src, const float* __restrict__ neigh,
                 const float* __restrict__ e0, float* __restrict__ et,
                 const _Float16* __restrict__ Weh, const float* __restrict__ p_logit,
                 int E, uint32_t seed) {
  __shared__ float smem[8 * 16 * 32];
  int wave = (blockIdx.x * blockDim.x + threadIdx.x) >> 5;
  int ntiles = (E + 15) >> 4;
  if (wave >= ntiles) return;
  int lane = threadIdx.x & 31;
  float* tile = smem + ((threadIdx.x >> 5) & 7) * 512;
  int row = lane & 15, hi = lane >> 4;
  int base = wave << 4;
  int e = base + row;
  bool full = (base + 16 <= E);                       // wave-uniform
  bool eok = full || (e < E);
  CDParams q = cd_params(p_logit);
  int ko = hi << 4;

  float nv[16], rv[16];
  if (eok) {
    int s = src[e];
    const float* nrow = neigh + (size_t)s * 32;
    const float* rrow = et + (size_t)(e ^ 1) * 32;    // rev pair is inside this tile
    *(float4*)(nv + 0)  = *(const float4*)(nrow + ko);
    *(float4*)(nv + 4)  = *(const float4*)(nrow + ko + 4);
    *(float4*)(nv + 8)  = *(const float4*)(nrow + ko + 8);
    *(float4*)(nv + 12) = *(const float4*)(nrow + ko + 12);
    *(float4*)(rv + 0)  = *(const float4*)(rrow + ko);
    *(float4*)(rv + 4)  = *(const float4*)(rrow + ko + 4);
    *(float4*)(rv + 8)  = *(const float4*)(rrow + ko + 8);
    *(float4*)(rv + 12) = *(const float4*)(rrow + ko + 12);
  } else {
#pragma unroll
    for (int j = 0; j < 16; ++j) { nv[j] = 0.0f; rv[j] = 0.0f; }
  }

  v16h a;
#pragma unroll
  for (int j = 0; j < 16; ++j) {
    int k = ko + j;
    float m = nv[j] - rv[j];
    m = cdrop(m, q, seed, (uint32_t)e * 32u + (uint32_t)k);
    a[j] = (_Float16)m;
  }

  int n = row, kb = hi << 4;
  v16h b0 = *(const v16h*)(Weh + (n)      * 32 + kb);
  v16h b1 = *(const v16h*)(Weh + (n + 16) * 32 + kb);
  v8f c0 = {}, c1 = {};
  c0 = WMMA_F16(a, b0, c0);
  c1 = WMMA_F16(a, b1, c1);

  if (full) {
    // stage GEMM tile in LDS; fuse residual + relu on coalesced b128 path
#pragma unroll
    for (int r = 0; r < 8; ++r) {
      int M = r + (hi << 3);
      tile[M * 32 + n]      = c0[r];
      tile[M * 32 + 16 + n] = c1[r];
    }
    LDS_FENCE();
    int orow = lane >> 1, half = lane & 1;
    const float* tr = tile + orow * 32 + half * 16;
    size_t off = (size_t)(base + orow) * 32 + (half << 4);
#pragma unroll
    for (int u = 0; u < 4; ++u) {
      float4 cv = *(const float4*)(tr + u * 4);
      float4 ev = *(const float4*)(e0 + off + u * 4);
      float4 o;
      o.x = fmaxf(ev.x + cv.x, 0.0f);
      o.y = fmaxf(ev.y + cv.y, 0.0f);
      o.z = fmaxf(ev.z + cv.z, 0.0f);
      o.w = fmaxf(ev.w + cv.w, 0.0f);
      *(float4*)(et + off + u * 4) = o;
    }
  } else {
#pragma unroll
    for (int r = 0; r < 8; ++r) {
      int M = r + (hi << 3);
      int ee = base + M;
      if (ee < E) {
        size_t off = (size_t)ee * 32 + n;
        float v0 = fmaxf(e0[off]      + c0[r], 0.0f);
        float v1 = fmaxf(e0[off + 16] + c1[r], 0.0f);
        et[off]      = v0;
        et[off + 16] = v1;
      }
    }
  }
}

// ---------------- readout: out = relu(W_last @ CD(concat(n_feat, ff))) ----------------

__global__ __launch_bounds__(256)
void readout(const float* __restrict__ nf, const float* __restrict__ ff,
             const _Float16* __restrict__ Wlh, const float* __restrict__ p_logit,
             float* __restrict__ out, int N, uint32_t seed) {
  __shared__ float smem[8 * 16 * 64];                 // 4 KB per wave
  int wave = (blockIdx.x * blockDim.x + threadIdx.x) >> 5;
  int ntiles = (N + 15) >> 4;
  if (wave >= ntiles) return;
  int lane = threadIdx.x & 31;
  float* tile = smem + ((threadIdx.x >> 5) & 7) * 1024;
  int row = lane & 15, hi = lane >> 4;
  int base = wave << 4;
  int node = base + row;
  bool full = (base + 16 <= N);                       // wave-uniform
  bool ok = full || (node < N);
  CDParams q = cd_params(p_logit);
  int ko = hi << 4;

  // z[k]: k<15 -> n_feat, 15<=k<47 -> ff, else 0.  K padded to 64 -> two A fragments.
  v16h a0, a1;
#pragma unroll
  for (int j = 0; j < 16; ++j) {
    int k0 = ko + j;
    int k1 = 32 + k0;
    float x0 = 0.0f, x1 = 0.0f;
    if (ok) {
      x0 = (k0 < 15) ? nf[(size_t)node * 15 + k0] : ff[(size_t)node * 32 + (k0 - 15)];
      x0 = cdrop(x0, q, seed, (uint32_t)node * 64u + (uint32_t)k0);
      if (k1 < 47) {
        x1 = ff[(size_t)node * 32 + (k1 - 15)];
        x1 = cdrop(x1, q, seed, (uint32_t)node * 64u + (uint32_t)k1);
      }
    }
    a0[j] = (_Float16)x0;
    a1[j] = (_Float16)x1;
  }

  int n = row, kb = hi << 4;
  if (full) {
#pragma unroll
    for (int blk = 0; blk < 4; ++blk) {
      int o = (blk << 4) + n;
      v16h blo = *(const v16h*)(Wlh + o * 64 + kb);        // K 0..31
      v16h bhi = *(const v16h*)(Wlh + o * 64 + 32 + kb);   // K 32..63
      v8f c = {};
      c = WMMA_F16(a0, blo, c);
      c = WMMA_F16(a1, bhi, c);
#pragma unroll
      for (int r = 0; r < 8; ++r) {
        int M = r + (hi << 3);
        tile[M * 64 + (blk << 4) + n] = fmaxf(c[r], 0.0f);
      }
    }
    LDS_FENCE();
    int orow = lane >> 1, half = lane & 1;
    const float* tr = tile + orow * 64 + half * 32;
    float* op = out + (size_t)(base + orow) * 64 + (half << 5);
#pragma unroll
    for (int u = 0; u < 8; ++u)
      *(float4*)(op + u * 4) = *(const float4*)(tr + u * 4);
  } else {
#pragma unroll
    for (int blk = 0; blk < 4; ++blk) {
      int o = (blk << 4) + n;
      v16h blo = *(const v16h*)(Wlh + o * 64 + kb);
      v16h bhi = *(const v16h*)(Wlh + o * 64 + 32 + kb);
      v8f c = {};
      c = WMMA_F16(a0, blo, c);
      c = WMMA_F16(a1, bhi, c);
#pragma unroll
      for (int r = 0; r < 8; ++r) {
        int M = r + (hi << 3);
        int nd = base + M;
        if (nd < N)
          out[(size_t)nd * 64 + (blk << 4) + n] = fmaxf(c[r], 0.0f);
      }
    }
  }
}

// ---------------- host launcher ----------------

extern "C" void kernel_launch(void* const* d_in, const int* in_sizes, int n_in,
                              void* d_out, int out_size, void* d_ws, size_t ws_size,
                              hipStream_t stream) {
  const float* nf     = (const float*)d_in[0];  // [N,15]
  const float* ef     = (const float*)d_in[1];  // [E,5]
  const float* Wi     = (const float*)d_in[2];  // [32,20]
  const float* We     = (const float*)d_in[3];  // [32,32]
  const float* Wl     = (const float*)d_in[4];  // [64,47]
  const float* p_init = (const float*)d_in[5];
  const float* p_eupd = (const float*)d_in[6];
  const float* p_last = (const float*)d_in[7];
  const int*   src    = (const int*)d_in[8];    // [E]
  const int*   dst    = (const int*)d_in[9];    // [E]
  int N = in_sizes[0] / 15;
  int E = in_sizes[1] / 5;
  (void)n_in; (void)out_size; (void)ws_size;

  // workspace carve-out (256B aligned): e0[E,32] + et[E,32] + neigh[N,32] + f16 weights
  char* w = (char*)d_ws;
  auto carve = [&](size_t bytes) -> char* {
    char* p = w; w += (bytes + 255) & ~(size_t)255; return p;
  };
  float*    e0    = (float*)carve((size_t)E * 32 * sizeof(float));
  float*    et    = (float*)carve((size_t)E * 32 * sizeof(float));
  float*    neigh = (float*)carve((size_t)N * 32 * sizeof(float));
  _Float16* Wih   = (_Float16*)carve(32 * 32 * sizeof(_Float16));
  _Float16* Weh   = (_Float16*)carve(32 * 32 * sizeof(_Float16));
  _Float16* Wlh   = (_Float16*)carve(64 * 64 * sizeof(_Float16));

  convert_weights<<<1, 256, 0, stream>>>(Wi, We, Wl, Wih, Weh, Wlh);

  int etiles  = (E + 15) / 16;
  int eblocks = (etiles + 7) / 8;          // 8 waves (tiles) per 256-thread block
  init_msg<<<eblocks, 256, 0, stream>>>(nf, ef, src, Wih, p_init, e0, et, E, 0u);

  int nElems  = N * 32;
  int zblocks = (nElems + 255) / 256;
  int sblocks = (E * 8 + 255) / 256;
  for (int i = 0; i < 3; ++i) {
    zero_f32<<<zblocks, 256, 0, stream>>>(neigh, nElems);
    scatter_add32<<<sblocks, 256, 0, stream>>>(et, dst, neigh, E);
    edge_update<<<eblocks, 256, 0, stream>>>(src, neigh, e0, et, Weh, p_eupd, E,
                                             (uint32_t)(1 + i));
  }

  // final segment_sum into neigh (= ff), then readout
  zero_f32<<<zblocks, 256, 0, stream>>>(neigh, nElems);
  scatter_add32<<<sblocks, 256, 0, stream>>>(et, dst, neigh, E);

  int ntilesN = (N + 15) / 16;
  int nblocks = (ntilesN + 7) / 8;
  readout<<<nblocks, 256, 0, stream>>>(nf, neigh, Wlh, p_last, (float*)d_out, N, 4u);
}
